// LSTMModel_without_theta_32014686224943
// MI455X (gfx1250) — matrix operand order repaired and checked by reading, not compiled, not run on hardware
//
#include <hip/hip_runtime.h>
#include <stdint.h>

#ifndef __has_builtin
#define __has_builtin(x) 0
#endif
#define USE_TDM 1

// ---------------------------------------------------------------------------
// raw (trivially-copyable) vector types for LDS/global block moves + WMMA
// ---------------------------------------------------------------------------
typedef __attribute__((ext_vector_type(16))) __bf16       v16bf;
typedef __attribute__((ext_vector_type(8)))  float        v8f;
typedef __attribute__((ext_vector_type(4)))  unsigned int u32x4;
typedef __attribute__((ext_vector_type(8)))  int          i32x8;
typedef __attribute__((ext_vector_type(4)))  int          i32x4;
typedef __attribute__((ext_vector_type(4)))  unsigned int u4raw;

#define B_   64
#define T_   2048
#define H_   256
#define NBLK 8

// ---------------- workspace layout (bytes) ----------------
#define WS_CNT   0                               // grid-barrier counter (4B, padded)
#define WS_H0    256                             // ushort[2][64][256] h0 double buffer
#define WS_H1    (WS_H0 + 2*B_*H_*2)             // ushort[2][64][256] h1 double buffer
#define WS_HMEAN (WS_H1 + 2*B_*H_*2)             // float [64][256]
#define WS_A1    (WS_HMEAN + B_*H_*4)            // float [64][256]
#define WS_A2    (WS_A1 + B_*H_*4)               // float [64][256]
#define WS_WT0   (WS_A2 + B_*H_*4)               // bf16 [4][256][256]  layer0 W slices (N-major)
#define WS_WT1   (WS_WT0 + 4*256*256*2)          // bf16 [4][256][512]  layer1 [Wi1;Wh1] slices

__device__ inline unsigned short f2bf(float f) {          // RNE float->bf16
  unsigned u = __float_as_uint(f);
  u += 0x7FFFu + ((u >> 16) & 1u);
  return (unsigned short)(u >> 16);
}
__device__ inline float sigm(float x) { return 1.f / (1.f + __expf(-x)); }
__device__ inline float tanh_(float x) {
  float e = __expf(-2.f * fabsf(x));
  float r = (1.f - e) / (1.f + e);
  return x >= 0.f ? r : -r;
}

// ---------------------------------------------------------------------------
// Prep: zero barrier/h buffers; build transposed bf16 weight slices.
// Layer-l slice for WG j holds local cols c=0..255 (gate g=c/64, within=c%64,
// global z column = g*256 + j*64 + within), K-contiguous (N-major => WMMA B
// fragments are two 16B LDS reads per lane).
// ---------------------------------------------------------------------------
__global__ void prep_k(const float* __restrict__ Wh0, const float* __restrict__ Wi1,
                       const float* __restrict__ Wh1, char* __restrict__ ws) {
  int tid  = blockIdx.x * blockDim.x + threadIdx.x;
  int nthr = gridDim.x * blockDim.x;
  if (tid == 0) *(unsigned*)(ws + WS_CNT) = 0u;
  unsigned short* hz = (unsigned short*)(ws + WS_H0);       // h0 + h1 contiguous
  for (int i = tid; i < 2 * 2 * B_ * H_; i += nthr) hz[i] = 0;

  unsigned short* wt0 = (unsigned short*)(ws + WS_WT0);
  for (int i = tid; i < 4 * 256 * 256; i += nthr) {
    int k = i & 255, c = (i >> 8) & 255, jw = i >> 16;
    int col = (c >> 6) * 256 + jw * 64 + (c & 63);
    wt0[i] = f2bf(Wh0[k * 1024 + col]);
  }
  unsigned short* wt1 = (unsigned short*)(ws + WS_WT1);
  for (int i = tid; i < 4 * 256 * 512; i += nthr) {
    int k = i & 511, c = (i >> 9) & 255, jw = i >> 17;
    int col = (c >> 6) * 256 + jw * 64 + (c & 63);
    float v = (k < 256) ? Wi1[k * 1024 + col] : Wh1[(k - 256) * 1024 + col];
    wt1[i] = f2bf(v);
  }
}

// ---------------------------------------------------------------------------
// Persistent pipelined 2-layer LSTM scan.
//  LDS map: [ Wslice bf16 256*K | union{ A bf16 64*K  /  Z f32 64*256 } ]
//  layer0: 128K + 64K = 192K ; layer1: 256K + 64K = 320K (max per WG).
// ---------------------------------------------------------------------------
template <int LAYER, int K>
__device__ void lstm_run(const float* __restrict__ x, const float* __restrict__ Wi0,
                         const float* __restrict__ b0, const float* __restrict__ b1,
                         char* __restrict__ ws, char* smem, int j) {
  const int tid  = threadIdx.x;
  const int lane = tid & 31;
  const int wid  = tid >> 5;
  const int lm   = lane & 15;
  const int hi16 = (lane >> 4) & 1;

  unsigned short* Wlds = (unsigned short*)smem;
  unsigned short* Alds = (unsigned short*)(smem + 256 * K * 2);  // aliases zbuf
  float*          zbuf = (float*)(smem + 256 * K * 2);

  unsigned short* h0buf = (unsigned short*)(ws + WS_H0);
  unsigned short* h1buf = (unsigned short*)(ws + WS_H1);
  unsigned*       cnt   = (unsigned*)(ws + WS_CNT);
  const unsigned short* gW = (const unsigned short*)
      (ws + (LAYER ? (WS_WT1 + j * 256 * 512 * 2) : (WS_WT0 + j * 256 * 256 * 2)));

  // ---- resident weight slice: global -> LDS via Tensor Data Mover ----
#if USE_TDM && __has_builtin(__builtin_amdgcn_tensor_load_to_lds)
  if (tid < 32) {  // one wave issues the DMA (TDM ignores EXEC)
    uint64_t ga     = (uint64_t)(uintptr_t)gW;
    uint32_t ldsoff = (uint32_t)(uintptr_t)Wlds;   // low 32b of LDS aperture addr = byte offset
    u32x4 g0; i32x8 g1;
    i32x4 g2 = {0, 0, 0, 0}, g3 = {0, 0, 0, 0};
    i32x8 g4 = {0, 0, 0, 0, 0, 0, 0, 0};           // unused D# group (this toolchain: 6-arg form)
    g0[0] = 1u;                                    // count=1, user descriptor
    g0[1] = ldsoff;                                // lds_addr
    g0[2] = (uint32_t)ga;                          // global_addr[31:0]
    g0[3] = (uint32_t)((ga >> 32) & 0x01FFFFFFu) | (2u << 30);  // addr[56:32], type=2
    g1[0] = (int)(1u << 16);                       // data_size=1 (2 bytes/elem)
    g1[1] = (int)(((unsigned)K & 0xFFFFu) << 16);  // tensor_dim0 = K (row length)
    g1[2] = (int)(256u << 16);                     // tensor_dim1 = 256 rows
    g1[3] = (int)(((unsigned)K & 0xFFFFu) << 16);  // tile_dim0 = K
    g1[4] = 256;                                   // tile_dim1 = 256, tile_dim2 = 0
    g1[5] = K;                                     // tensor_dim0_stride = K
    g1[6] = 0; g1[7] = 0;
    __builtin_amdgcn_tensor_load_to_lds(g0, g1, g2, g3, g4, 0);
    __builtin_amdgcn_s_wait_tensorcnt(0);
  }
#else
  for (int i = tid; i < 256 * K / 8; i += 256)
    ((u4raw*)Wlds)[i] = ((const u4raw*)gW)[i];
#endif

  {  // h_prev = 0 for the first GEMM
    u4raw z = {0, 0, 0, 0};
    for (int i = tid; i < B_ * K / 8; i += 256) ((u4raw*)Alds)[i] = z;
  }
  __syncthreads();

  float creg[16], hsum[16];
#pragma unroll
  for (int q = 0; q < 16; ++q) { creg[q] = 0.f; hsum[q] = 0.f; }
  const int b_row = tid >> 2;
  const int n0    = (tid & 3) << 4;
  unsigned short* myH = LAYER ? h1buf : h0buf;

  for (int s = 0; s <= T_; ++s) {
    const bool active = LAYER ? (s >= 1) : (s < T_);
    const int  t      = LAYER ? (s - 1) : s;

    v8f acc[8];
    if (active) {
#pragma unroll
      for (int i = 0; i < 8; ++i)
#pragma unroll
        for (int r = 0; r < 8; ++r) acc[i][r] = 0.f;

      for (int kk = 0; kk < K / 32; ++kk) {
        union { v16bf v; u4raw q[2]; } a[4], bb[2];
#pragma unroll
        for (int m = 0; m < 4; ++m) {  // A 16x32 bf16: K0-7/K16-23 (lo lanes), K8-15/K24-31 (hi)
          const u4raw* p = (const u4raw*)(Alds + (16 * m + lm) * K + kk * 32 + hi16 * 8);
          a[m].q[0] = p[0]; a[m].q[1] = p[2];
        }
#pragma unroll
        for (int nn = 0; nn < 2; ++nn) {  // B 32x16 bf16: lane=col, K0-7/K16-23 (lo lanes), K8-15/K24-31 (hi)
          const u4raw* p = (const u4raw*)(Wlds + (16 * (2 * wid + nn) + lm) * K + kk * 32 + hi16 * 8);
          bb[nn].q[0] = p[0]; bb[nn].q[1] = p[2];
        }
#pragma unroll
        for (int m = 0; m < 4; ++m)
#pragma unroll
          for (int nn = 0; nn < 2; ++nn)
            acc[m * 2 + nn] = __builtin_amdgcn_wmma_f32_16x16x32_bf16(
                false, a[m].v, false, bb[nn].v, (short)0, acc[m * 2 + nn], false, false);
      }
    }
    __syncthreads();  // everyone done reading Alds (it aliases zbuf)
    if (active) {
#pragma unroll
      for (int m = 0; m < 4; ++m)
#pragma unroll
        for (int nn = 0; nn < 2; ++nn) {
          int col = 16 * (2 * wid + nn) + lm;
#pragma unroll
          for (int r = 0; r < 8; ++r)
            zbuf[(16 * m + r + hi16 * 8) * 256 + col] = acc[m * 2 + nn][r];
        }
    }
    __syncthreads();

    if (active) {  // gates: this thread owns (b_row, n0..n0+15); c & running mean in registers
      const int slot = s & 1;
#pragma unroll 4
      for (int q = 0; q < 16; ++q) {
        int   n  = n0 + q;
        float zi = zbuf[b_row * 256 + n];
        float zf = zbuf[b_row * 256 + 64 + n];
        float zg = zbuf[b_row * 256 + 128 + n];
        float zo = zbuf[b_row * 256 + 192 + n];
        int   ci = j * 64 + n;  // column within each gate block
        if (LAYER == 0) {       // D=1 input projection is a rank-1 update + bias
          float xv = x[b_row * T_ + t];
          zi += xv * Wi0[ci]       + b0[ci];
          zf += xv * Wi0[256 + ci] + b0[256 + ci];
          zg += xv * Wi0[512 + ci] + b0[512 + ci];
          zo += xv * Wi0[768 + ci] + b0[768 + ci];
        } else {
          zi += b1[ci]; zf += b1[256 + ci]; zg += b1[512 + ci]; zo += b1[768 + ci];
        }
        float iv = sigm(zi), fv = sigm(zf), gv = tanh_(zg), ov = sigm(zo);
        creg[q] = fv * creg[q] + iv * gv;
        float hv = ov * tanh_(creg[q]);
        myH[(slot * B_ + b_row) * H_ + j * 64 + n] = f2bf(hv);
        if (LAYER) hsum[q] += hv;
      }
    }

    // ---- release h slices, grid-wide superstep barrier ----
    __threadfence();
    __syncthreads();
#if __has_builtin(__builtin_amdgcn_s_cluster_barrier)
    __builtin_amdgcn_s_cluster_barrier();  // NOP unless dispatched as a cluster
#endif
    if (tid == 0) {
      __hip_atomic_fetch_add(cnt, 1u, __ATOMIC_ACQ_REL, __HIP_MEMORY_SCOPE_AGENT);
      while (__hip_atomic_load(cnt, __ATOMIC_ACQUIRE, __HIP_MEMORY_SCOPE_AGENT) <
             (unsigned)(NBLK * (s + 1)))
        __builtin_amdgcn_s_sleep(1);
    }
    __syncthreads();
    __threadfence();

    // ---- stage next A = h (layer0) or [h0_t ; h1_prev] (layer1) into LDS ----
    if (s < T_) {
      const int slot = s & 1;
      if (LAYER == 0) {
        const u4raw* src = (const u4raw*)(h0buf + slot * B_ * H_);
        u4raw*       dst = (u4raw*)Alds;
        for (int i = tid; i < B_ * H_ / 8; i += 256) dst[i] = src[i];
      } else {
        const u4raw* s0 = (const u4raw*)(h0buf + slot * B_ * H_);
        const u4raw* s1 = (const u4raw*)(h1buf + slot * B_ * H_);
        u4raw*       dst = (u4raw*)Alds;
        for (int i = tid; i < B_ * 512 / 8; i += 256) {
          int r = i >> 6, cu = i & 63;  // 64 u4raw per 512-col row; first 32 = h0, next 32 = h1
          dst[i] = (cu < 32) ? s0[r * 32 + cu] : s1[r * 32 + cu - 32];
        }
      }
    }
    __syncthreads();
  }

  if (LAYER == 1) {  // write time-mean of h1 (kept in registers all along)
    float* hmean = (float*)(ws + WS_HMEAN);
#pragma unroll
    for (int q = 0; q < 16; ++q)
      hmean[b_row * H_ + j * 64 + n0 + q] = hsum[q] * (1.f / (float)T_);
  }
}

__global__ __launch_bounds__(256) void lstm_persist(const float* x, const float* Wi0,
                                                    const float* b0, const float* b1,
                                                    char* ws) {
  extern __shared__ __align__(16) char smem[];
  int wg = blockIdx.x, j = wg & 3;
  if (wg < 4) lstm_run<0, 256>(x, Wi0, b0, b1, ws, smem, j);
  else        lstm_run<1, 512>(x, Wi0, b0, b1, ws, smem, j);
}

// ---------------------------------------------------------------------------
// Tiny MLP head (64x256 @ 256x256, ELU) — negligible FLOPs, plain fp32.
// ---------------------------------------------------------------------------
__global__ void dense_elu_k(const float* __restrict__ in, const float* __restrict__ W,
                            const float* __restrict__ bias, float* __restrict__ out) {
  int idx = blockIdx.x * blockDim.x + threadIdx.x;  // 64*256
  int b = idx >> 8, n = idx & 255;
  float s = bias[n];
  for (int k = 0; k < 256; ++k) s += in[b * 256 + k] * W[k * 256 + n];
  out[idx] = s > 0.f ? s : (__expf(s) - 1.f);
}
__global__ void head_k(const float* __restrict__ a2, const float* __restrict__ Wo,
                       const float* __restrict__ bo, float* __restrict__ out) {
  int b = threadIdx.x;  // 64
  float s = bo[0];
  for (int k = 0; k < 256; ++k) s += a2[b * 256 + k] * Wo[k];
  out[b] = s;
}

extern "C" void kernel_launch(void* const* d_in, const int* in_sizes, int n_in,
                              void* d_out, int out_size, void* d_ws, size_t ws_size,
                              hipStream_t stream) {
  const float* x   = (const float*)d_in[0];
  const float* Wi0 = (const float*)d_in[1];
  const float* Wh0 = (const float*)d_in[2];
  const float* b0  = (const float*)d_in[3];
  const float* Wi1 = (const float*)d_in[4];
  const float* Wh1 = (const float*)d_in[5];
  const float* b1  = (const float*)d_in[6];
  const float* W1  = (const float*)d_in[7];
  const float* bl1 = (const float*)d_in[8];
  const float* W2  = (const float*)d_in[9];
  const float* bl2 = (const float*)d_in[10];
  const float* Wo  = (const float*)d_in[11];
  const float* bo  = (const float*)d_in[12];
  char* ws = (char*)d_ws;

  prep_k<<<128, 256, 0, stream>>>(Wh0, Wi1, Wh1, ws);
  lstm_persist<<<NBLK, 256, 320 * 1024, stream>>>(x, Wi0, b0, b1, ws);
  dense_elu_k<<<64, 256, 0, stream>>>((const float*)(ws + WS_HMEAN), W1, bl1,
                                      (float*)(ws + WS_A1));
  dense_elu_k<<<64, 256, 0, stream>>>((const float*)(ws + WS_A1), W2, bl2,
                                      (float*)(ws + WS_A2));
  head_k<<<1, 64, 0, stream>>>((const float*)(ws + WS_A2), Wo, bo, (float*)d_out);
}
